// SwellAttention_9586367005364
// MI455X (gfx1250) — compile-verified, matrix operand order checked by reading, not run on hardware
//
#include <hip/hip_runtime.h>
#include <hip/hip_bf16.h>

// MI455X / gfx1250, wave32. bf16 WMMA (v_wmma_f32_16x16x32_bf16) everywhere.
// Pipeline: prep (fp32->bf16 x + transposed weights) -> qkv GEMM (16x64/wave,
// K-loop register double-buffered) -> fused flash attention (4 q-tile
// waves/block, V loads overlapped with softmax) -> proj GEMM (16x64/wave).

typedef __attribute__((ext_vector_type(16))) __bf16 v16bf;
typedef __attribute__((ext_vector_type(8)))  __bf16 v8bf;
typedef __attribute__((ext_vector_type(8)))  float  v8f;

#define BATCH 32
#define NPATCH 576
#define CH 768
#define NH 12
#define HD 64
#define C3 2304
#define GS 24           // sqrt(NPATCH)
#define MSZ 47          // 2*GS-1
#define CURD 12         // masking distance (mask_ratio=0.5)
#define MASKV (-100.0f)

// ---- fragment helpers ------------------------------------------------------
// 16-bit A-fragment layout (CDNA5 ISA 7.12.2), 16x32 tile:
//   lane = hl*16 + r ; elements j=0..7  -> K = hl*8 + j
//                       elements j=8..15 -> K = 16 + hl*8 + (j-8)
// B-fragment for column n=r uses the same striping over rows of B^T, so both
// loaders are two contiguous 16-byte loads.

__device__ inline v16bf frag_bf16(const __bf16* base /*row start + k0*/, int hl) {
    v8bf lo = *reinterpret_cast<const v8bf*>(base + hl * 8);
    v8bf hi = *reinterpret_cast<const v8bf*>(base + 16 + hl * 8);
    return __builtin_shufflevector(lo, hi, 0,1,2,3,4,5,6,7,8,9,10,11,12,13,14,15);
}

__device__ inline v8f wmma_bf16(v16bf a, v16bf b, v8f c) {
    return __builtin_amdgcn_wmma_f32_16x16x32_bf16(false, a, false, b, (short)0, c,
                                                   false, false);
}

__device__ inline float redmax16(float v) {
#pragma unroll
    for (int off = 8; off >= 1; off >>= 1) v = fmaxf(v, __shfl_xor(v, off, 32));
    return v;
}
__device__ inline float redsum16(float v) {
#pragma unroll
    for (int off = 8; off >= 1; off >>= 1) v += __shfl_xor(v, off, 32);
    return v;
}

// ---- kernel 0: x -> bf16, weights -> bf16 transposed -----------------------
__global__ void prep_convert(const float* __restrict__ x,
                             const float* __restrict__ w_qkv,
                             const float* __restrict__ w_proj,
                             __bf16* __restrict__ xb,
                             __bf16* __restrict__ wqkvT,
                             __bf16* __restrict__ wprojT) {
    const int n_x   = BATCH * NPATCH * CH;   // 14,155,776
    const int n_qkv = CH * C3;               //  1,769,472
    const int n_prj = CH * CH;               //    589,824
    int i = blockIdx.x * blockDim.x + threadIdx.x;
    if (i < n_x) {
        xb[i] = (__bf16)x[i];
    } else if (i < n_x + n_qkv) {
        int j = i - n_x;
        int k = j / C3, n = j % C3;
        wqkvT[(size_t)n * CH + k] = (__bf16)w_qkv[j];
    } else if (i < n_x + n_qkv + n_prj) {
        int j = i - n_x - n_qkv;
        int k = j / CH, n = j % CH;
        wprojT[(size_t)n * CH + k] = (__bf16)w_proj[j];
    }
}

// ---- double-buffered 16x64-per-wave GEMM core ------------------------------
// acc[c] += A(16xCH) @ B^T rows (cg*64 + c*16 .. +15)
__device__ inline void gemm_16x64(const __bf16* __restrict__ arow,
                                  const __bf16* __restrict__ brow0,
                                  const __bf16* __restrict__ brow1,
                                  const __bf16* __restrict__ brow2,
                                  const __bf16* __restrict__ brow3,
                                  int hl, v8f acc[4]) {
    const __bf16* brow[4] = {brow0, brow1, brow2, brow3};
    v16bf af = frag_bf16(arow, hl);
    v16bf bf[4];
#pragma unroll
    for (int c = 0; c < 4; ++c) bf[c] = frag_bf16(brow[c], hl);

#pragma unroll
    for (int k0 = 0; k0 < CH; k0 += 32) {
        v16bf afn;
        v16bf bfn[4];
        if (k0 + 32 < CH) {                 // prefetch next stage into regs
            afn = frag_bf16(arow + k0 + 32, hl);
#pragma unroll
            for (int c = 0; c < 4; ++c) bfn[c] = frag_bf16(brow[c] + k0 + 32, hl);
        }
#pragma unroll
        for (int c = 0; c < 4; ++c) acc[c] = wmma_bf16(af, bf[c], acc[c]);
        if (k0 + 32 < CH) {
            af = afn;
#pragma unroll
            for (int c = 0; c < 4; ++c) bf[c] = bfn[c];
        }
    }
}

// ---- kernel 1: qkv = x @ w_qkv + b ; 16x64 tile per wave -------------------
// scatter to q,k [B,H,N,64] (q pre-scaled by 1/8) and vT [B,H,64,N]
__global__ void __launch_bounds__(128)
qkv_gemm(const __bf16* __restrict__ xb,
         const __bf16* __restrict__ wT,
         const float* __restrict__ bias,
         __bf16* __restrict__ q,
         __bf16* __restrict__ k,
         __bf16* __restrict__ vT) {
    const int lane = threadIdx.x & 31;
    const int wave = threadIdx.x >> 5;
    const int r  = lane & 15;
    const int hl = lane >> 4;
    const int mt = blockIdx.x;                 // 0..1151 row tiles (B*N/16)
    const int cg = blockIdx.y * 4 + wave;      // 0..35 col groups of 64
    const __bf16* arow = xb + (size_t)(mt * 16 + r) * CH;
    const __bf16* bbase = wT + (size_t)(cg * 64 + r) * CH;

    v8f acc[4] = {{}, {}, {}, {}};
    gemm_16x64(arow, bbase, bbase + 16 * CH, bbase + 32 * CH, bbase + 48 * CH,
               hl, acc);

    const int which = (cg * 64) / CH;          // 0=q 1=k 2=v (wave-uniform)
    const int h     = ((cg * 64) % CH) >> 6;   // head (wave-uniform)
    const int b     = mt / (NPATCH / 16);      // tile never crosses batch
    const int nb    = (mt % (NPATCH / 16)) * 16;
    const size_t bh = (size_t)b * NH + h;
#pragma unroll
    for (int c = 0; c < 4; ++c) {
        const int d  = c * 16 + r;             // 0..63 within head
        const float bb = bias[cg * 64 + c * 16 + r];
#pragma unroll
        for (int g = 0; g < 8; ++g) {
            const int n = nb + g + 8 * hl;
            const float val = acc[c][g] + bb;
            if (which == 0)
                q[(bh * NPATCH + n) * HD + d] = (__bf16)(val * 0.125f);
            else if (which == 1)
                k[(bh * NPATCH + n) * HD + d] = (__bf16)val;
            else
                vT[(bh * HD + d) * NPATCH + n] = (__bf16)val;
        }
    }
}

// ---- kernel 2: fused flash attention; 4 q-tile waves per block -------------
__global__ void __launch_bounds__(128)
attn_fused(const __bf16* __restrict__ q,
           const __bf16* __restrict__ kk,
           const __bf16* __restrict__ vT,
           const float* __restrict__ rel_table,   // [2209, 12]
           __bf16* __restrict__ ao) {             // [B, N, C] bf16
    const int lane = threadIdx.x & 31;
    const int wv = threadIdx.x >> 5;
    const int r  = lane & 15;
    const int hl = lane >> 4;
    const int qt = blockIdx.x * 4 + wv;     // 0..35
    const int h  = blockIdx.y;              // 0..11
    const int b  = blockIdx.z;              // 0..31
    const size_t bh = (size_t)b * NH + h;

    const __bf16* qbase = q + (bh * NPATCH + qt * 16 + r) * HD;
    v16bf aq0 = frag_bf16(qbase + 0, hl);
    v16bf aq1 = frag_bf16(qbase + 32, hl);

    v8f o[4] = {{}, {}, {}, {}};
    float m[8], l[8];
    int qy[8], qx[8];
#pragma unroll
    for (int g = 0; g < 8; ++g) {
        m[g] = -3.0e38f;
        l[g] = 0.0f;
        const int qr = qt * 16 + g + 8 * hl;
        qy[g] = qr / GS;
        qx[g] = qr % GS;
    }

    __shared__ __align__(16) __bf16 pbuf[4][16][32];

    for (int kb = 0; kb < NPATCH / 32; ++kb) {
        // ---- S = Q K^T for a 16x32 key block (4 WMMA) ----
        const __bf16* kbase = kk + (bh * NPATCH + kb * 32 + r) * HD;
        __builtin_prefetch(kbase + 32 * HD, 0, 3);
        v16bf bk00 = frag_bf16(kbase + 0, hl);
        v16bf bk01 = frag_bf16(kbase + 32, hl);
        v16bf bk10 = frag_bf16(kbase + 16 * HD + 0, hl);
        v16bf bk11 = frag_bf16(kbase + 16 * HD + 32, hl);
        v8f s[2] = {{}, {}};
        s[0] = wmma_bf16(aq0, bk00, s[0]);
        s[0] = wmma_bf16(aq1, bk01, s[0]);
        s[1] = wmma_bf16(aq0, bk10, s[1]);
        s[1] = wmma_bf16(aq1, bk11, s[1]);

        // ---- issue V-fragment loads early: consumed only after softmax ----
        v16bf bv[4];
#pragma unroll
        for (int t2 = 0; t2 < 4; ++t2) {
            const __bf16* vbase = vT + (bh * HD + t2 * 16 + r) * NPATCH + kb * 32;
            __builtin_prefetch(vbase + 32, 0, 3);
            bv[t2] = frag_bf16(vbase, hl);
        }

        // ---- relative-position bias + window mask (analytic) ----
#pragma unroll
        for (int t = 0; t < 2; ++t) {
            const int key = kb * 32 + t * 16 + r;
            const int ky = key / GS, kx = key % GS;
#pragma unroll
            for (int g = 0; g < 8; ++g) {
                const int dy = qy[g] - ky;
                const int dx = qx[g] - kx;
                const int ay = dy < 0 ? -dy : dy;
                const int ax = dx < 0 ? -dx : dx;
                const int cheb = ay > ax ? ay : ax;
                const int idx = (dy + GS - 1) * MSZ + (dx + GS - 1);
                float add = rel_table[idx * NH + h];
                if (cheb > CURD) add += MASKV;
                s[t][g] += add;
            }
        }

        // ---- online softmax update ----
        float c[8];
#pragma unroll
        for (int g = 0; g < 8; ++g) {
            float rm = redmax16(fmaxf(s[0][g], s[1][g]));
            float mn = fmaxf(m[g], rm);
            c[g] = __expf(m[g] - mn);
            m[g] = mn;
        }
#pragma unroll
        for (int g = 0; g < 8; ++g) {
            s[0][g] = __expf(s[0][g] - m[g]);
            s[1][g] = __expf(s[1][g] - m[g]);
            float rs = redsum16(s[0][g] + s[1][g]);
            l[g] = l[g] * c[g] + rs;
        }
#pragma unroll
        for (int t2 = 0; t2 < 4; ++t2)
#pragma unroll
            for (int g = 0; g < 8; ++g) o[t2][g] *= c[g];

        // ---- re-shape P from C/D layout to a 16x32 bf16 A-fragment via LDS --
        __syncthreads();
#pragma unroll
        for (int t = 0; t < 2; ++t)
#pragma unroll
            for (int g = 0; g < 8; ++g)
                pbuf[wv][g + 8 * hl][t * 16 + r] = (__bf16)s[t][g];
        __syncthreads();
        v16bf pf = frag_bf16(&pbuf[wv][r][0], hl);

        // ---- O += P V (4 WMMA, V fragments already resident) ----
#pragma unroll
        for (int t2 = 0; t2 < 4; ++t2) o[t2] = wmma_bf16(pf, bv[t2], o[t2]);
    }

    // finalize: O /= l, store to [B, N, H*64+d] bf16
#pragma unroll
    for (int g = 0; g < 8; ++g) l[g] = 1.0f / l[g];
#pragma unroll
    for (int t2 = 0; t2 < 4; ++t2)
#pragma unroll
        for (int g = 0; g < 8; ++g) {
            const int qr = qt * 16 + g + 8 * hl;
            ao[((size_t)b * NPATCH + qr) * CH + h * HD + t2 * 16 + r] =
                (__bf16)(o[t2][g] * l[g]);
        }
}

// ---- kernel 3: out = attn_out @ w_proj + b_proj ; 16x64 tile per wave ------
__global__ void __launch_bounds__(128)
proj_gemm(const __bf16* __restrict__ ao,
          const __bf16* __restrict__ wT,
          const float* __restrict__ bias,
          float* __restrict__ out) {
    const int lane = threadIdx.x & 31;
    const int wave = threadIdx.x >> 5;
    const int r  = lane & 15;
    const int hl = lane >> 4;
    const int mt = blockIdx.x;                 // 0..1151
    const int cg = blockIdx.y * 4 + wave;      // 0..11 col groups of 64
    const __bf16* arow = ao + (size_t)(mt * 16 + r) * CH;
    const __bf16* bbase = wT + (size_t)(cg * 64 + r) * CH;

    v8f acc[4] = {{}, {}, {}, {}};
    gemm_16x64(arow, bbase, bbase + 16 * CH, bbase + 32 * CH, bbase + 48 * CH,
               hl, acc);
#pragma unroll
    for (int c = 0; c < 4; ++c) {
        const int col = cg * 64 + c * 16 + r;
        const float bb = bias[col];
#pragma unroll
        for (int g = 0; g < 8; ++g) {
            const int rowg = mt * 16 + g + 8 * hl;
            out[(size_t)rowg * CH + col] = acc[c][g] + bb;
        }
    }
}

// ---- launcher --------------------------------------------------------------
extern "C" void kernel_launch(void* const* d_in, const int* in_sizes, int n_in,
                              void* d_out, int out_size, void* d_ws, size_t ws_size,
                              hipStream_t stream) {
    const float* x         = (const float*)d_in[0];
    const float* w_qkv     = (const float*)d_in[1];
    const float* b_qkv     = (const float*)d_in[2];
    const float* rel_table = (const float*)d_in[3];
    const float* w_proj    = (const float*)d_in[4];
    const float* b_proj    = (const float*)d_in[5];
    float* out = (float*)d_out;

    // workspace layout (bytes)
    char* ws = (char*)d_ws;
    const size_t sz_xb    = (size_t)BATCH * NPATCH * CH * 2;     // 28.3 MB
    const size_t sz_wqkvT = (size_t)C3 * CH * 2;                 //  3.54 MB
    const size_t sz_wprjT = (size_t)CH * CH * 2;                 //  1.18 MB
    const size_t sz_head  = (size_t)BATCH * NH * NPATCH * HD * 2;// 28.3 MB
    __bf16* xb    = (__bf16*)(ws);
    __bf16* wqkvT = (__bf16*)(ws + sz_xb);
    __bf16* wprjT = (__bf16*)(ws + sz_xb + sz_wqkvT);
    char* p = ws + sz_xb + sz_wqkvT + sz_wprjT;
    __bf16* qb  = (__bf16*)(p);
    __bf16* kb  = (__bf16*)(p + sz_head);
    __bf16* vTb = (__bf16*)(p + 2 * sz_head);
    __bf16* aob = (__bf16*)(p + 3 * sz_head);

    // 0) x + weights -> bf16 (weights transposed)
    {
        const int total = BATCH * NPATCH * CH + CH * C3 + CH * CH;
        prep_convert<<<(total + 255) / 256, 256, 0, stream>>>(x, w_qkv, w_proj,
                                                              xb, wqkvT, wprjT);
    }
    // 1) qkv GEMM: row tiles x (3C/256) col-group blocks, 4 waves/block
    {
        dim3 grid(BATCH * NPATCH / 16, C3 / 256);
        qkv_gemm<<<grid, 128, 0, stream>>>(xb, wqkvT, b_qkv, qb, kb, vTb);
    }
    // 2) fused attention: 4 q-tile waves per block
    {
        dim3 grid(NPATCH / 64, NH, BATCH);
        attn_fused<<<grid, 128, 0, stream>>>(qb, kb, vTb, rel_table, aob);
    }
    // 3) projection GEMM
    {
        dim3 grid(BATCH * NPATCH / 16, CH / 256);
        proj_gemm<<<grid, 128, 0, stream>>>(aob, wprjT, b_proj, out);
    }
}
